// SimplifiedMambaBlock_5540507812270
// MI455X (gfx1250) — compile-verified, matrix operand order checked by reading, not run on hardware
//
#include <hip/hip_runtime.h>
#include <hip/hip_bf16.h>
#include <math.h>

typedef __attribute__((ext_vector_type(16))) _Float16 v16h;
typedef __attribute__((ext_vector_type(8)))  _Float16 v8h_t;
typedef __attribute__((ext_vector_type(8)))  float    v8f;

#define D_MODEL  1024
#define D_STATE  32
#define D_CONV   4
#define D_INNER  2048
#define DT_RANK  64
#define BATCH    2
#define SEQLEN   512
#define NTOK     (BATCH * SEQLEN)        // 1024
#define XPROJ_N  (DT_RANK + 2 * D_STATE) // 128
#define LN_EPS   1e-5f

// ---------------------------------------------------------------------------
// Helpers
// ---------------------------------------------------------------------------

// Load one lane's 16-f16 fragment of a 16x32 (row x K) tile for WMMA.
// CDNA5 16-bit A/B layout: lane = half*16 + r holds row `r`, elements
// 0..7 -> K = k0 + half*8 + e, elements 8..15 -> K = k0 + 16 + half*8 + e.
// Both runs are contiguous -> two global_load_b128 per lane.
__device__ __forceinline__ v16h load_frag16(const _Float16* __restrict__ base,
                                            int ld, int row, int k0, int half) {
  const _Float16* p = base + (size_t)row * ld + k0 + half * 8;
  v8h_t lo = *(const v8h_t*)(p);
  v8h_t hi = *(const v8h_t*)(p + 16);
  v16h r;
#pragma unroll
  for (int i = 0; i < 8; ++i) { r[i] = lo[i]; r[i + 8] = hi[i]; }
  return r;
}

// ---------------------------------------------------------------------------
// WMMA GEMM: C[M,N] = A[M,K] @ W[N,K]^T  (+ optional fp32 residual)
// A, W row-major f16; C fp32. One wave computes a 32(M) x 64(N) strip:
// 2 A-fragments x 4 B-fragments -> 8 v_wmma per 32-deep k-chunk, so every
// B b128-load pair feeds two WMMAs (~21 flop/byte from L0).
// NO bounds checks: launch geometry must tile M and N exactly
//   (gridDim.y*32 == M, gridDim.x * (blockDim.x/32) * 64 == N).
// Branch-free k-loop keeps EXEC all-1s around WMMA and lets the compiler
// clause the b128 loads and keep accumulators pinned in even VGPRs.
// ---------------------------------------------------------------------------
#define GEMM_TM 2
#define GEMM_TN 4

__global__ void wmma_gemm_f16(const _Float16* __restrict__ A,
                              const _Float16* __restrict__ W,
                              const float* __restrict__ resid,
                              float* __restrict__ C,
                              int M, int N, int K) {
  const int wave   = threadIdx.x >> 5;
  const int lane   = threadIdx.x & 31;
  const int m0     = blockIdx.y * (16 * GEMM_TM);
  const int n_base = (blockIdx.x * (blockDim.x >> 5) + wave) * (16 * GEMM_TN);
  const int r    = lane & 15;
  const int half = lane >> 4;

  const _Float16* __restrict__ Arow = A + (size_t)(m0 + r) * K;

  v8f acc[GEMM_TM][GEMM_TN] = {};
  for (int k0 = 0; k0 < K; k0 += 32) {
    if (k0 + 32 < K)  // speculative prefetch of next A chunk (global_prefetch_b8)
      __builtin_prefetch(Arow + k0 + 32, 0, 1);
    v16h a[GEMM_TM];
#pragma unroll
    for (int mt = 0; mt < GEMM_TM; ++mt)
      a[mt] = load_frag16(A, K, m0 + 16 * mt + r, k0, half);
#pragma unroll
    for (int t = 0; t < GEMM_TN; ++t) {
      v16h b = load_frag16(W, K, n_base + 16 * t + r, k0, half);
#pragma unroll
      for (int mt = 0; mt < GEMM_TM; ++mt)
        acc[mt][t] = __builtin_amdgcn_wmma_f32_16x16x32_f16(
            false, a[mt], false, b, (short)0, acc[mt][t], false, false);
    }
  }

  // C/D layout: VGPR v, lane l -> row = m0 + 16*mt + 8*(l>>4) + v, col = n0 + (l&15)
  if (resid) {
#pragma unroll
    for (int mt = 0; mt < GEMM_TM; ++mt)
#pragma unroll
      for (int t = 0; t < GEMM_TN; ++t) {
        const int col = n_base + 16 * t + r;
#pragma unroll
        for (int v = 0; v < 8; ++v) {
          const int row = m0 + 16 * mt + 8 * half + v;
          C[(size_t)row * N + col] = acc[mt][t][v] + resid[(size_t)row * N + col];
        }
      }
  } else {
#pragma unroll
    for (int mt = 0; mt < GEMM_TM; ++mt)
#pragma unroll
      for (int t = 0; t < GEMM_TN; ++t) {
        const int col = n_base + 16 * t + r;
#pragma unroll
        for (int v = 0; v < 8; ++v) {
          const int row = m0 + 16 * mt + 8 * half + v;
          C[(size_t)row * N + col] = acc[mt][t][v];
        }
      }
  }
}

// ---------------------------------------------------------------------------
// LayerNorm over D_MODEL, output f16 (GEMM operand precision)
// ---------------------------------------------------------------------------
__global__ void layernorm_f16(const float* __restrict__ x,
                              const float* __restrict__ gamma,
                              const float* __restrict__ beta,
                              _Float16* __restrict__ out) {
  __shared__ float s1[256], s2[256];
  const int t = blockIdx.x;
  const float* xp = x + (size_t)t * D_MODEL;
  float a = 0.f, b = 0.f;
  for (int i = threadIdx.x; i < D_MODEL; i += 256) {
    float v = xp[i];
    a += v; b += v * v;
  }
  s1[threadIdx.x] = a; s2[threadIdx.x] = b;
  __syncthreads();
  for (int off = 128; off > 0; off >>= 1) {
    if ((int)threadIdx.x < off) {
      s1[threadIdx.x] += s1[threadIdx.x + off];
      s2[threadIdx.x] += s2[threadIdx.x + off];
    }
    __syncthreads();
  }
  const float mu  = s1[0] * (1.f / D_MODEL);
  const float var = s2[0] * (1.f / D_MODEL) - mu * mu;
  const float inv = rsqrtf(var + LN_EPS);
  for (int i = threadIdx.x; i < D_MODEL; i += 256)
    out[(size_t)t * D_MODEL + i] = (_Float16)((xp[i] - mu) * inv * gamma[i] + beta[i]);
}

// ---------------------------------------------------------------------------
// Elementwise kernels
// ---------------------------------------------------------------------------
__global__ void cvt_f32_f16(const float* __restrict__ in,
                            _Float16* __restrict__ out, int n) {
  int i = blockIdx.x * blockDim.x + threadIdx.x;
  if (i < n) out[i] = (_Float16)in[i];
}

// causal depthwise conv (k=4) + bias + SiLU; reads u-half of xz [tok, 4096]
__global__ void conv_silu(const float* __restrict__ xz,
                          const float* __restrict__ cw,
                          const float* __restrict__ cb,
                          float* __restrict__ u,
                          _Float16* __restrict__ u_h) {
  int i = blockIdx.x * blockDim.x + threadIdx.x;     // over NTOK*D_INNER
  if (i >= NTOK * D_INNER) return;
  const int d   = i % D_INNER;
  const int tok = i / D_INNER;
  const int l   = tok % SEQLEN;
  const int b   = tok / SEQLEN;
  float acc = cb[d];
#pragma unroll
  for (int j = 0; j < D_CONV; ++j) {
    const int ls = l - (D_CONV - 1) + j;
    if (ls >= 0)
      acc += xz[(size_t)(b * SEQLEN + ls) * (2 * D_INNER) + d] * cw[d * D_CONV + j];
  }
  const float s = acc / (1.f + __expf(-acc));        // SiLU
  u[i]   = s;
  u_h[i] = (_Float16)s;
}

__global__ void slice_dt_f16(const float* __restrict__ x_dbl,
                             _Float16* __restrict__ dt_h) {
  int i = blockIdx.x * blockDim.x + threadIdx.x;     // over NTOK*DT_RANK
  if (i >= NTOK * DT_RANK) return;
  const int tok = i / DT_RANK, c = i % DT_RANK;
  dt_h[i] = (_Float16)x_dbl[(size_t)tok * XPROJ_N + c];
}

__global__ void softplus_bias(float* __restrict__ delta,
                              const float* __restrict__ b_dt) {
  int i = blockIdx.x * blockDim.x + threadIdx.x;     // over NTOK*D_INNER
  if (i >= NTOK * D_INNER) return;
  const int d = i % D_INNER;
  const float v = delta[i] + b_dt[d];
  delta[i] = (v > 20.f) ? v : log1pf(__expf(v));
}

// ---------------------------------------------------------------------------
// Selective scan: one wave32 per (batch, channel); lane = state index n.
// h[n] = exp(delta*A[n]) * h[n] + delta*B[n]*u ; y = sum_n h[n]*C[n]
// Fuses skip (u*D), SiLU(z) gate and f16 conversion of y.
// ---------------------------------------------------------------------------
__global__ void selective_scan(const float* __restrict__ xz,     // [NTOK, 4096] (z half)
                               const float* __restrict__ u,      // [NTOK, D_INNER]
                               const float* __restrict__ x_dbl,  // [NTOK, 128]
                               const float* __restrict__ delta,  // [NTOK, D_INNER]
                               const float* __restrict__ A_log,  // [D_INNER, 32]
                               const float* __restrict__ Dp,     // [D_INNER]
                               _Float16* __restrict__ y_h) {     // [NTOK, D_INNER]
  const int wave = threadIdx.x >> 5;
  const int lane = threadIdx.x & 31;
  const int ch   = blockIdx.x * (blockDim.x >> 5) + wave;  // 0..BATCH*D_INNER-1
  const int b = ch / D_INNER;
  const int d = ch % D_INNER;

  const float An = -__expf(A_log[(size_t)d * D_STATE + lane]);
  const float Dd = Dp[d];
  float h = 0.f;

  for (int l = 0; l < SEQLEN; ++l) {
    const int tok = b * SEQLEN + l;
    const float dlt = delta[(size_t)tok * D_INNER + d];   // wave-broadcast
    const float uu  = u[(size_t)tok * D_INNER + d];       // wave-broadcast
    const float Bn  = x_dbl[(size_t)tok * XPROJ_N + DT_RANK + lane];
    const float Cn  = x_dbl[(size_t)tok * XPROJ_N + DT_RANK + D_STATE + lane];

    const float dA = __expf(dlt * An);                    // v_exp_f32 (TRANS)
    h = dA * h + dlt * Bn * uu;

    float yp = h * Cn;                                    // reduce over 32 lanes
#pragma unroll
    for (int off = 16; off > 0; off >>= 1)
      yp += __shfl_xor(yp, off, 32);

    if (lane == 0) {
      const float z = xz[(size_t)tok * (2 * D_INNER) + D_INNER + d];
      const float g = z / (1.f + __expf(-z));             // SiLU gate
      y_h[(size_t)tok * D_INNER + d] = (_Float16)((yp + uu * Dd) * g);
    }
  }
}

// ---------------------------------------------------------------------------
// Host-side orchestration
// ---------------------------------------------------------------------------
extern "C" void kernel_launch(void* const* d_in, const int* in_sizes, int n_in,
                              void* d_out, int out_size, void* d_ws, size_t ws_size,
                              hipStream_t stream) {
  const float* x      = (const float*)d_in[0];
  const float* gamma  = (const float*)d_in[1];
  const float* beta   = (const float*)d_in[2];
  const float* W_in   = (const float*)d_in[3];
  const float* conv_w = (const float*)d_in[4];
  const float* conv_b = (const float*)d_in[5];
  const float* W_x    = (const float*)d_in[6];
  const float* W_dt   = (const float*)d_in[7];
  const float* b_dt   = (const float*)d_in[8];
  const float* A_log  = (const float*)d_in[9];
  const float* Dp     = (const float*)d_in[10];
  const float* W_out  = (const float*)d_in[11];
  float* out = (float*)d_out;

  // Workspace carve-up (256B aligned)
  char* p = (char*)d_ws;
  auto carve = [&](size_t bytes) {
    char* r = p;
    p += (bytes + 255) & ~(size_t)255;
    return r;
  };
  _Float16* W_in_h  = (_Float16*)carve((size_t)2 * D_INNER * D_MODEL * 2);
  _Float16* W_x_h   = (_Float16*)carve((size_t)XPROJ_N * D_INNER * 2);
  _Float16* W_dt_h  = (_Float16*)carve((size_t)D_INNER * DT_RANK * 2);
  _Float16* W_out_h = (_Float16*)carve((size_t)D_MODEL * D_INNER * 2);
  _Float16* xn_h    = (_Float16*)carve((size_t)NTOK * D_MODEL * 2);
  float*    xz      = (float*)   carve((size_t)NTOK * 2 * D_INNER * 4);
  float*    u       = (float*)   carve((size_t)NTOK * D_INNER * 4);
  _Float16* u_h     = (_Float16*)carve((size_t)NTOK * D_INNER * 2);
  float*    x_dbl   = (float*)   carve((size_t)NTOK * XPROJ_N * 4);
  _Float16* dt_h    = (_Float16*)carve((size_t)NTOK * DT_RANK * 2);
  float*    delta   = (float*)   carve((size_t)NTOK * D_INNER * 4);
  _Float16* y_h     = (_Float16*)carve((size_t)NTOK * D_INNER * 2);

  const int T = 256;
  const int WAVE_STRIP = 16 * GEMM_TN;   // 64 cols per wave
  const int M_TILE     = 16 * GEMM_TM;   // 32 rows per block

  // Weight conversion fp32 -> f16
  cvt_f32_f16<<<(2 * D_INNER * D_MODEL + T - 1) / T, T, 0, stream>>>(W_in, W_in_h, 2 * D_INNER * D_MODEL);
  cvt_f32_f16<<<(XPROJ_N * D_INNER + T - 1) / T, T, 0, stream>>>(W_x, W_x_h, XPROJ_N * D_INNER);
  cvt_f32_f16<<<(D_INNER * DT_RANK + T - 1) / T, T, 0, stream>>>(W_dt, W_dt_h, D_INNER * DT_RANK);
  cvt_f32_f16<<<(D_MODEL * D_INNER + T - 1) / T, T, 0, stream>>>(W_out, W_out_h, D_MODEL * D_INNER);

  // LayerNorm -> f16
  layernorm_f16<<<NTOK, T, 0, stream>>>(x, gamma, beta, xn_h);

  // in_proj: xz = xn @ W_in^T   [1024, 4096]
  // 8 waves/block * 64 = 512 cols/block -> grid.x = 8; grid.y = 1024/32 = 32
  {
    dim3 g((2 * D_INNER) / (8 * WAVE_STRIP), NTOK / M_TILE);
    wmma_gemm_f16<<<g, 256, 0, stream>>>(xn_h, W_in_h, nullptr, xz, NTOK, 2 * D_INNER, D_MODEL);
  }

  // depthwise conv + SiLU
  conv_silu<<<(NTOK * D_INNER + T - 1) / T, T, 0, stream>>>(xz, conv_w, conv_b, u, u_h);

  // x_proj: x_dbl = u @ W_x^T   [1024, 128]
  // 2 waves/block (64 threads) * 64 = 128 cols -> grid.x = 1 covers N exactly
  {
    dim3 g(1, NTOK / M_TILE);
    wmma_gemm_f16<<<g, 64, 0, stream>>>(u_h, W_x_h, nullptr, x_dbl, NTOK, XPROJ_N, D_INNER);
  }

  // dt slice -> f16
  slice_dt_f16<<<(NTOK * DT_RANK + T - 1) / T, T, 0, stream>>>(x_dbl, dt_h);

  // dt_proj: delta_pre = dt @ W_dt^T   [1024, 2048]
  {
    dim3 g(D_INNER / (8 * WAVE_STRIP), NTOK / M_TILE);
    wmma_gemm_f16<<<g, 256, 0, stream>>>(dt_h, W_dt_h, nullptr, delta, NTOK, D_INNER, DT_RANK);
  }

  // delta = softplus(delta_pre + b_dt)
  softplus_bias<<<(NTOK * D_INNER + T - 1) / T, T, 0, stream>>>(delta, b_dt);

  // selective scan (wave per channel, lane per state), fused skip+gate -> y_h f16
  selective_scan<<<(BATCH * D_INNER) / 8, T, 0, stream>>>(
      xz, u, x_dbl, delta, A_log, Dp, y_h);

  // out_proj + residual: out = y @ W_out^T + x   [1024, 1024]
  {
    dim3 g(D_MODEL / (8 * WAVE_STRIP), NTOK / M_TILE);
    wmma_gemm_f16<<<g, 256, 0, stream>>>(y_h, W_out_h, x, out, NTOK, D_MODEL, D_INNER);
  }
}
